// AttentionGNN_73632919323215
// MI455X (gfx1250) — matrix operand order repaired and checked by reading, not compiled, not run on hardware
//
#include <hip/hip_runtime.h>
#include <math.h>

typedef __attribute__((ext_vector_type(16))) _Float16 v16h;
typedef __attribute__((ext_vector_type(8)))  float    v8f;

#define BATCH 8
#define SEQ   2048
#define DIM   256
#define INV_TEMP 0.0625f   // 1/sqrt(256)

union FragU { v16h h; uint4 q[2]; };

// A-fragment (16x32 f16): lane m=lane&15, h=lane>>4 reads halfs [8h..8h+7] and [16+8h..16+8h+7]
__device__ __forceinline__ v16h load_fragA(const _Float16* row, int lh) {
  const uint4* p = (const uint4*)row;
  FragU f;
  f.q[0] = p[lh];
  f.q[1] = p[2 + lh];
  return f.h;
}

// B-fragment (32x16 f16 stored transposed in LDS as Bt[n][32]): lane reads halfs [8h..8h+7] and [16+8h..16+8h+7]
__device__ __forceinline__ v16h load_fragB(const _Float16* row, int lh) {
  const uint4* p = (const uint4*)row;
  FragU f;
  f.q[0] = p[lh];
  f.q[1] = p[2 + lh];
  return f.h;
}

// CDNA5 async global->LDS copy (16B per lane), tracked by ASYNCcnt.
__device__ __forceinline__ void async_copy_b128(const void* gsrc, void* ldst) {
  unsigned lds_off = (unsigned)(size_t)ldst;   // low 32 bits = wave-relative LDS offset
  asm volatile("global_load_async_to_lds_b128 %0, %1, off"
               :: "v"(lds_off), "v"(gsrc)
               : "memory");
}

__device__ __forceinline__ void wait_async() {
  asm volatile("s_wait_asynccnt 0x0" ::: "memory");
}

// ---------------------------------------------------------------------------
// Tensor Data Mover: 2D f16 tile load, descriptor per CDNA5 ISA ch.8.
// This toolchain exposes the 6-arg builtin:
//   (u32x4 g0, i32x8 g1, i32x4 g2, i32x4 g3, i32x8 extra, i32 cpol)
// ---------------------------------------------------------------------------
#if __has_builtin(__builtin_amdgcn_tensor_load_to_lds)
#define HAVE_TDM 1
typedef unsigned int u32x4 __attribute__((ext_vector_type(4)));
typedef int i32x8 __attribute__((ext_vector_type(8)));
typedef int i32x4 __attribute__((ext_vector_type(4)));

__device__ __forceinline__ void tdm_load_2d_f16(const void* gsrc, void* ldst,
                                                unsigned tileX, unsigned tileY,
                                                unsigned strideElems) {
  unsigned long long ga = (unsigned long long)(size_t)gsrc;
  unsigned lds = (unsigned)(size_t)ldst;
  u32x4 g0;
  g0[0] = 1u;                                                // count=1 (valid user D#)
  g0[1] = lds;                                               // lds_addr
  g0[2] = (unsigned)ga;                                      // global_addr[31:0]
  g0[3] = (unsigned)((ga >> 32) & 0x01FFFFFFu) | (2u << 30); // addr[56:32] | type=2
  i32x8 g1;
  g1[0] = (int)(1u << 16);                                   // data_size=1 (2 bytes)
  g1[1] = (int)((tileX & 0xFFFFu) << 16);                    // tensor_dim0[15:0]
  g1[2] = (int)((tileX >> 16) & 0xFFFFu) |
          (int)((tileY & 0xFFFFu) << 16);                    // dim0 hi | tensor_dim1 lo
  g1[3] = (int)((tileY >> 16) & 0xFFFFu) |
          (int)((tileX & 0xFFFFu) << 16);                    // dim1 hi | tile_dim0
  g1[4] = (int)(tileY & 0xFFFFu);                            // tile_dim1 (tile_dim2=0)
  g1[5] = (int)strideElems;                                  // tensor_dim0_stride[31:0]
  g1[6] = 0;
  g1[7] = 0;
  i32x4 gz4 = {0, 0, 0, 0};
  i32x8 gz8 = {0, 0, 0, 0, 0, 0, 0, 0};
  __builtin_amdgcn_tensor_load_to_lds(g0, g1, gz4, gz4, gz8, 0);
}

__device__ __forceinline__ void wait_tensor() {
  __builtin_amdgcn_s_wait_tensorcnt(0);
}
#else
#define HAVE_TDM 0
#endif

// ---------------------------------------------------------------------------
// Kernel 1: q/k/v = text @ W + b  (A f32->f16 convert, bias folded into C)
// ---------------------------------------------------------------------------
__global__ __launch_bounds__(256) void qkv_kernel(const float* __restrict__ text,
                                                  const float* __restrict__ W,
                                                  const float* __restrict__ bias,
                                                  _Float16* __restrict__ out) {
  __shared__ alignas(16) _Float16 As[64][32];
  __shared__ alignas(16) _Float16 Bt[128][32];

  const int tid  = threadIdx.x;
  const int lane = tid & 31;
  const int wave = tid >> 5;
  const int mSub = wave & 3;
  const int nSub = wave >> 2;
  const int ln = lane & 15;
  const int lh = lane >> 4;
  const int rowBlock = blockIdx.x * 64;
  const int nBlock   = blockIdx.y * 128;

  v8f acc[4];
#pragma unroll
  for (int t = 0; t < 4; ++t) {
    float bv = bias[nBlock + nSub * 64 + t * 16 + ln];
#pragma unroll
    for (int r = 0; r < 8; ++r) acc[t][r] = bv;
  }

  for (int kk = 0; kk < DIM; kk += 32) {
    {
      int r = tid >> 2, cs = (tid & 3) * 8;
      const float* src = text + (size_t)(rowBlock + r) * DIM + kk + cs;
#pragma unroll
      for (int j = 0; j < 8; ++j) As[r][cs + j] = (_Float16)src[j];
    }
    {
      int kd = tid >> 3, ns = (tid & 7) * 16;
      const float* src = W + (size_t)(kk + kd) * DIM + nBlock + ns;
#pragma unroll
      for (int j = 0; j < 16; ++j) Bt[ns + j][kd] = (_Float16)src[j];
    }
    __syncthreads();
    v16h aF = load_fragA(&As[mSub * 16 + ln][0], lh);
#pragma unroll
    for (int t = 0; t < 4; ++t) {
      v16h bF = load_fragB(&Bt[nSub * 64 + t * 16 + ln][0], lh);
      acc[t] = __builtin_amdgcn_wmma_f32_16x16x32_f16(false, aF, false, bF,
                                                      (short)0, acc[t], false, false);
    }
    __syncthreads();
  }

  const int row0 = rowBlock + mSub * 16 + 8 * lh;
#pragma unroll
  for (int t = 0; t < 4; ++t) {
    int col = nBlock + nSub * 64 + t * 16 + ln;
#pragma unroll
    for (int r = 0; r < 8; ++r)
      out[(size_t)(row0 + r) * DIM + col] = (_Float16)acc[t][r];
  }
}

// ---------------------------------------------------------------------------
// Kernel 2: scores = q @ k^T * (1/16) -> f32 scratch (new_adj region of d_out)
// Both tiles verbatim f16 copies -> async global->LDS path.
// ---------------------------------------------------------------------------
__global__ __launch_bounds__(256) void scores_kernel(const _Float16* __restrict__ q,
                                                     const _Float16* __restrict__ k,
                                                     float* __restrict__ scores) {
  __shared__ alignas(16) _Float16 As[64][32];
  __shared__ alignas(16) _Float16 Bt[128][32];

  const int tid  = threadIdx.x;
  const int lane = tid & 31;
  const int wave = tid >> 5;
  const int mSub = wave & 3;
  const int nSub = wave >> 2;
  const int ln = lane & 15;
  const int lh = lane >> 4;
  const int b = blockIdx.z;
  const int iBlock = blockIdx.x * 64;
  const int jBlock = blockIdx.y * 128;
  const _Float16* qb = q + (size_t)b * SEQ * DIM;
  const _Float16* kb = k + (size_t)b * SEQ * DIM;
  float* sb = scores + (size_t)b * SEQ * SEQ;

  const int ar = tid >> 2, ac = (tid & 3) * 8;   // A: one b128 per thread
  const int bn = tid >> 1, bc = (tid & 1) * 16;  // B: two b128 per thread

  v8f acc[4];
#pragma unroll
  for (int t = 0; t < 4; ++t)
#pragma unroll
    for (int r = 0; r < 8; ++r) acc[t][r] = 0.0f;

  for (int kk = 0; kk < DIM; kk += 32) {
    async_copy_b128(qb + (size_t)(iBlock + ar) * DIM + kk + ac, &As[ar][ac]);
    async_copy_b128(kb + (size_t)(jBlock + bn) * DIM + kk + bc, &Bt[bn][bc]);
    async_copy_b128(kb + (size_t)(jBlock + bn) * DIM + kk + bc + 8, &Bt[bn][bc + 8]);
    wait_async();
    __syncthreads();

    v16h aF = load_fragA(&As[mSub * 16 + ln][0], lh);
#pragma unroll
    for (int t = 0; t < 4; ++t) {
      v16h bF = load_fragB(&Bt[nSub * 64 + t * 16 + ln][0], lh);
      acc[t] = __builtin_amdgcn_wmma_f32_16x16x32_f16(false, aF, false, bF,
                                                      (short)0, acc[t], false, false);
    }
    __syncthreads();
  }

  const int row0 = iBlock + mSub * 16 + 8 * lh;
#pragma unroll
  for (int t = 0; t < 4; ++t) {
    int col = jBlock + nSub * 64 + t * 16 + ln;
#pragma unroll
    for (int r = 0; r < 8; ++r)
      sb[(size_t)(row0 + r) * SEQ + col] = acc[t][r] * INV_TEMP;
  }
}

// ---------------------------------------------------------------------------
// Kernel 3: row-wise softmax (2048 wide); f32 in -> f16 attn out
// ---------------------------------------------------------------------------
__global__ __launch_bounds__(256) void softmax_kernel(const float* __restrict__ scores,
                                                      _Float16* __restrict__ attn) {
  const size_t row = blockIdx.x;
  const float* s = scores + row * SEQ;
  _Float16* a = attn + row * SEQ;
  __shared__ float red[256];
  const int tid = threadIdx.x;

  float vals[8];
  float lmax = -3.4e38f;
#pragma unroll
  for (int j = 0; j < 8; ++j) {
    vals[j] = s[tid + j * 256];
    lmax = fmaxf(lmax, vals[j]);
  }
  red[tid] = lmax;
  __syncthreads();
  for (int st = 128; st > 0; st >>= 1) {
    if (tid < st) red[tid] = fmaxf(red[tid], red[tid + st]);
    __syncthreads();
  }
  float m = red[0];
  __syncthreads();

  float lsum = 0.0f;
#pragma unroll
  for (int j = 0; j < 8; ++j) {
    vals[j] = __expf(vals[j] - m);
    lsum += vals[j];
  }
  red[tid] = lsum;
  __syncthreads();
  for (int st = 128; st > 0; st >>= 1) {
    if (tid < st) red[tid] += red[tid + st];
    __syncthreads();
  }
  float inv = 1.0f / red[0];
#pragma unroll
  for (int j = 0; j < 8; ++j) a[tid + j * 256] = (_Float16)(vals[j] * inv);
}

// ---------------------------------------------------------------------------
// Kernel 4 (dominant, 137 GFLOP): new_adj = attn16 @ adj
// 64x256 block tile, 32x64 wave tile, double-buffered LDS.
// A tile staged by the Tensor Data Mover (wave 0 issues one 64x32 f16 2D
// descriptor per K-step); B tile staged f32->f16 through VALU.
// ---------------------------------------------------------------------------
__global__ __launch_bounds__(256) void newadj_kernel(const _Float16* __restrict__ attn,
                                                     const float* __restrict__ adj,
                                                     float* __restrict__ newadj) {
  __shared__ alignas(16) _Float16 As[2][64][32];    //  8 KB
  __shared__ alignas(16) _Float16 Bt[2][256][32];   // 32 KB

  const int tid  = threadIdx.x;
  const int lane = tid & 31;
  const int wave = tid >> 5;
  const int mW = wave & 1;   // 0..1 -> 32-row strip
  const int nW = wave >> 1;  // 0..3 -> 64-col strip
  const int ln = lane & 15;
  const int lh = lane >> 4;
  const int b = blockIdx.z;
  const int iBlock = blockIdx.x * 64;
  const int jBlock = blockIdx.y * 256;
  const _Float16* ab = attn + (size_t)b * SEQ * SEQ;
  const float* jb = adj + (size_t)b * SEQ * SEQ;
  float* ob = newadj + (size_t)b * SEQ * SEQ;

  const int ar = tid >> 2, ac = (tid & 3) * 8;      // A fallback: one b128 per thread
  const int bkd = tid >> 3, bns = (tid & 7) * 32;   // B: 32 halves per thread

  v8f acc[2][4];
#pragma unroll
  for (int s = 0; s < 2; ++s)
#pragma unroll
    for (int t = 0; t < 4; ++t)
#pragma unroll
      for (int r = 0; r < 8; ++r) acc[s][t][r] = 0.0f;

  // Prologue: stage K-slab 0 into buffer 0.
#if HAVE_TDM
  if (wave == 0)
    tdm_load_2d_f16(ab + (size_t)iBlock * SEQ, &As[0][0][0], 32, 64, SEQ);
#else
  async_copy_b128(ab + (size_t)(iBlock + ar) * SEQ + ac, &As[0][ar][ac]);
#endif
  {
    const float* src = jb + (size_t)bkd * SEQ + jBlock + bns;
#pragma unroll
    for (int j4 = 0; j4 < 8; ++j4) {
      float4 f = *(const float4*)(src + j4 * 4);
      Bt[0][bns + j4 * 4 + 0][bkd] = (_Float16)f.x;
      Bt[0][bns + j4 * 4 + 1][bkd] = (_Float16)f.y;
      Bt[0][bns + j4 * 4 + 2][bkd] = (_Float16)f.z;
      Bt[0][bns + j4 * 4 + 3][bkd] = (_Float16)f.w;
    }
  }
#if HAVE_TDM
  if (wave == 0) wait_tensor();
#else
  wait_async();
#endif
  __syncthreads();

  int cur = 0;
  for (int kk = 0; kk < SEQ; kk += 32) {
    const int nxt = cur ^ 1;
    if (kk + 32 < SEQ) {  // stage next slab while computing on current
#if HAVE_TDM
      if (wave == 0)
        tdm_load_2d_f16(ab + (size_t)iBlock * SEQ + (kk + 32), &As[nxt][0][0],
                        32, 64, SEQ);
#else
      async_copy_b128(ab + (size_t)(iBlock + ar) * SEQ + (kk + 32) + ac,
                      &As[nxt][ar][ac]);
#endif
      const float* src = jb + (size_t)(kk + 32 + bkd) * SEQ + jBlock + bns;
      __builtin_prefetch(src + (size_t)32 * SEQ, 0, 1);
#pragma unroll
      for (int j4 = 0; j4 < 8; ++j4) {
        float4 f = *(const float4*)(src + j4 * 4);
        Bt[nxt][bns + j4 * 4 + 0][bkd] = (_Float16)f.x;
        Bt[nxt][bns + j4 * 4 + 1][bkd] = (_Float16)f.y;
        Bt[nxt][bns + j4 * 4 + 2][bkd] = (_Float16)f.z;
        Bt[nxt][bns + j4 * 4 + 3][bkd] = (_Float16)f.w;
      }
    }

    v16h aF0 = load_fragA(&As[cur][mW * 32 + ln][0], lh);
    v16h aF1 = load_fragA(&As[cur][mW * 32 + 16 + ln][0], lh);
#pragma unroll
    for (int t = 0; t < 4; ++t) {
      v16h bF = load_fragB(&Bt[cur][nW * 64 + t * 16 + ln][0], lh);
      acc[0][t] = __builtin_amdgcn_wmma_f32_16x16x32_f16(false, aF0, false, bF,
                                                         (short)0, acc[0][t], false, false);
      acc[1][t] = __builtin_amdgcn_wmma_f32_16x16x32_f16(false, aF1, false, bF,
                                                         (short)0, acc[1][t], false, false);
    }

#if HAVE_TDM
    if (wave == 0) wait_tensor();
#else
    wait_async();
#endif
    __syncthreads();
    cur = nxt;
  }

#pragma unroll
  for (int s = 0; s < 2; ++s) {
    const int row0 = iBlock + mW * 32 + s * 16 + 8 * lh;
#pragma unroll
    for (int t = 0; t < 4; ++t) {
      int col = jBlock + nW * 64 + t * 16 + ln;
#pragma unroll
      for (int r = 0; r < 8; ++r)
        ob[(size_t)(row0 + r) * SEQ + col] = acc[s][t][r];
    }
  }
}

// ---------------------------------------------------------------------------
// Kernel 5: output = new_adj(->f16) @ v16 -> f32 d_out output region
// ---------------------------------------------------------------------------
__global__ __launch_bounds__(256) void output_kernel(const float* __restrict__ newadj,
                                                     const _Float16* __restrict__ v,
                                                     float* __restrict__ out) {
  __shared__ alignas(16) _Float16 As[64][32];
  __shared__ alignas(16) _Float16 Bt[128][32];

  const int tid  = threadIdx.x;
  const int lane = tid & 31;
  const int wave = tid >> 5;
  const int mSub = wave & 3;
  const int nSub = wave >> 2;
  const int ln = lane & 15;
  const int lh = lane >> 4;
  const int b = blockIdx.z;
  const int iBlock = blockIdx.x * 64;
  const int nBlock = blockIdx.y * 128;
  const float* nb = newadj + (size_t)b * SEQ * SEQ;
  const _Float16* vb = v + (size_t)b * SEQ * DIM;
  float* ob = out + (size_t)b * SEQ * DIM;

  v8f acc[4];
#pragma unroll
  for (int t = 0; t < 4; ++t)
#pragma unroll
    for (int r = 0; r < 8; ++r) acc[t][r] = 0.0f;

  for (int kk = 0; kk < SEQ; kk += 32) {
    {
      int r = tid >> 2, cs = (tid & 3) * 8;
      const float* src = nb + (size_t)(iBlock + r) * SEQ + kk + cs;
#pragma unroll
      for (int j = 0; j < 8; ++j) As[r][cs + j] = (_Float16)src[j];
    }
    {
      int kd = tid >> 3, ns = (tid & 7) * 16;
      const _Float16* src = vb + (size_t)(kk + kd) * DIM + nBlock + ns;
#pragma unroll
      for (int j = 0; j < 16; ++j) Bt[ns + j][kd] = src[j];
    }
    __syncthreads();
    v16h aF = load_fragA(&As[mSub * 16 + ln][0], lh);
#pragma unroll
    for (int t = 0; t < 4; ++t) {
      v16h bF = load_fragB(&Bt[nSub * 64 + t * 16 + ln][0], lh);
      acc[t] = __builtin_amdgcn_wmma_f32_16x16x32_f16(false, aF, false, bF,
                                                      (short)0, acc[t], false, false);
    }
    __syncthreads();
  }

  const int row0 = iBlock + mSub * 16 + 8 * lh;
#pragma unroll
  for (int t = 0; t < 4; ++t) {
    int col = nBlock + nSub * 64 + t * 16 + ln;
#pragma unroll
    for (int r = 0; r < 8; ++r)
      ob[(size_t)(row0 + r) * DIM + col] = acc[t][r];
  }
}

// ---------------------------------------------------------------------------
extern "C" void kernel_launch(void* const* d_in, const int* in_sizes, int n_in,
                              void* d_out, int out_size, void* d_ws, size_t ws_size,
                              hipStream_t stream) {
  const float* text = (const float*)d_in[0];
  const float* adj  = (const float*)d_in[1];
  const float* Wq   = (const float*)d_in[2];
  const float* bq   = (const float*)d_in[3];
  const float* Wk   = (const float*)d_in[4];
  const float* bk   = (const float*)d_in[5];
  const float* Wv   = (const float*)d_in[6];
  const float* bv   = (const float*)d_in[7];

  float* out    = (float*)d_out;                    // [B,N,D]
  float* newadj = out + (size_t)BATCH * SEQ * DIM;  // [B,N,N] (also scores scratch)

  _Float16* q16 = (_Float16*)d_ws;                  // 8 MB
  _Float16* k16 = q16 + (size_t)BATCH * SEQ * DIM;  // 8 MB
  _Float16* v16 = k16 + (size_t)BATCH * SEQ * DIM;  // 8 MB
  _Float16* a16 = v16 + (size_t)BATCH * SEQ * DIM;  // 67 MB attn f16

  dim3 blk(256);

  dim3 gProj(BATCH * SEQ / 64, DIM / 128, 1);
  qkv_kernel<<<gProj, blk, 0, stream>>>(text, Wq, bq, q16);
  qkv_kernel<<<gProj, blk, 0, stream>>>(text, Wk, bk, k16);
  qkv_kernel<<<gProj, blk, 0, stream>>>(text, Wv, bv, v16);

  scores_kernel<<<dim3(SEQ / 64, SEQ / 128, BATCH), blk, 0, stream>>>(q16, k16, newadj);

  softmax_kernel<<<dim3(BATCH * SEQ), blk, 0, stream>>>(newadj, a16);

  newadj_kernel<<<dim3(SEQ / 64, SEQ / 256, BATCH), blk, 0, stream>>>(a16, adj, newadj);

  output_kernel<<<dim3(SEQ / 64, DIM / 128, BATCH), blk, 0, stream>>>(newadj, v16, out);
}
